// multihead_attention_25374666785428
// MI455X (gfx1250) — matrix operand order repaired and checked by reading, not compiled, not run on hardware
//
#include <hip/hip_runtime.h>

// ---------------------------------------------------------------------------
// CDNA5 (gfx1250) multi-head attention, bf16 WMMA path (v_wmma_f32_16x16x32_bf16)
// + async global->LDS staging (ASYNCcnt) and global_prefetch_b8 where useful.
// Problem: B=2, S=2048, H=1024, n_head=16, d=64
// ---------------------------------------------------------------------------

#define B_DIM 2
#define S_DIM 2048
#define H_DIM 1024
#define NHEAD 16
#define D_HEAD 64
#define MROWS (B_DIM * S_DIM)          // 4096
#define NEGV  (-4294967295.0f)

#if defined(__has_builtin)
#  if __has_builtin(__builtin_amdgcn_global_load_async_to_lds_b128)
#    define HAVE_ASYNC_LDS 1
#  endif
#  if __has_builtin(__builtin_amdgcn_s_wait_asynccnt)
#    define HAVE_WAIT_ASYNC_BUILTIN 1
#  endif
#endif

typedef __attribute__((ext_vector_type(16))) __bf16 v16bf;
typedef __attribute__((ext_vector_type(8)))  float  v8f;
typedef __attribute__((ext_vector_type(4)))  int    v4i;

typedef __attribute__((address_space(1))) v4i* gptr_v4i;   // global
typedef __attribute__((address_space(3))) v4i* lptr_v4i;   // LDS

union Frag {            // 16 bf16 per lane = 8 VGPRs = two 16B chunks
    v16bf v;
    uint4 q[2];
};

__device__ __forceinline__ unsigned short f2bf(float f) {
    unsigned u = __float_as_uint(f);
    u += 0x7FFFu + ((u >> 16) & 1u);     // round-to-nearest-even
    return (unsigned short)(u >> 16);
}
__device__ __forceinline__ float bf2f(unsigned short h) {
    return __uint_as_float(((unsigned)h) << 16);
}

__device__ __forceinline__ v8f wmma_bf16(const Frag& a, const Frag& b, v8f c) {
    // D(16x16 f32) = A(16x32 bf16) * B(32x16 bf16) + C
    return __builtin_amdgcn_wmma_f32_16x16x32_bf16(
        false, a.v, false, b.v, (short)0, c, false, false);
}

__device__ __forceinline__ void wait_asynccnt0() {
#if defined(HAVE_WAIT_ASYNC_BUILTIN)
    __builtin_amdgcn_s_wait_asynccnt(0);
#else
    asm volatile("s_wait_asynccnt 0" ::: "memory");
#endif
}

// ---------------------------------------------------------------------------
// GEMM: Y = relu(X[ M x K ] * W[ K x N ] + bias)
//  mode 0: write bf16, head-major [head][row][d]   (row = b*S+s)
//  mode 1: write fp32 Y + resid (residual add), row-major [row][N]
// Block: 128 threads (4 waves), 64x64 tile, wave owns 32x32.
// ---------------------------------------------------------------------------
__global__ __launch_bounds__(128)
void gemm_bias_relu(const float* __restrict__ X, const float* __restrict__ W,
                    const float* __restrict__ bias, int M, int Kd, int N,
                    int mode,
                    unsigned short* __restrict__ outBf,
                    float* __restrict__ outF, const float* __restrict__ resid)
{
    __shared__ unsigned short Xl[64][32];   // A tile, row-major, bf16
    __shared__ unsigned short Wt[64][32];   // B tile, transposed [n][k], bf16

    const int tid  = threadIdx.x;
    const int lane = tid & 31;
    const int wave = tid >> 5;
    const int half = lane >> 4;
    const int l16  = lane & 15;
    const int rowBase = blockIdx.x * 64;
    const int colBase = blockIdx.y * 64;
    const int wr = (wave >> 1) * 32;        // wave row offset in tile
    const int wc = (wave & 1) * 32;         // wave col offset in tile

    v8f acc[2][2];
    acc[0][0] = {}; acc[0][1] = {}; acc[1][0] = {}; acc[1][1] = {};

    for (int k0 = 0; k0 < Kd; k0 += 32) {
        // prefetch next K-step tiles into cache (global_prefetch_b8)
        if (k0 + 32 < Kd) {
            __builtin_prefetch(X + (size_t)(rowBase + (tid >> 1)) * Kd + k0 + 32 + (tid & 1) * 16, 0, 1);
            __builtin_prefetch(W + (size_t)(k0 + 32 + (tid >> 2)) * N + colBase + (tid & 3) * 16, 0, 1);
        }
        // stage X tile (64x32 fp32 -> bf16): 512 float4 loads / 128 threads
        #pragma unroll
        for (int i = 0; i < 4; ++i) {
            int idx = tid + i * 128;
            int r = idx >> 3;
            int c = (idx & 7) << 2;
            float4 f = *(const float4*)(X + (size_t)(rowBase + r) * Kd + k0 + c);
            Xl[r][c + 0] = f2bf(f.x); Xl[r][c + 1] = f2bf(f.y);
            Xl[r][c + 2] = f2bf(f.z); Xl[r][c + 3] = f2bf(f.w);
        }
        // stage W tile transposed (32x64 fp32 -> Wt[n][k] bf16)
        #pragma unroll
        for (int i = 0; i < 4; ++i) {
            int idx = tid + i * 128;
            int kr = idx >> 4;
            int n  = (idx & 15) << 2;
            float4 f = *(const float4*)(W + (size_t)(k0 + kr) * N + colBase + n);
            Wt[n + 0][kr] = f2bf(f.x); Wt[n + 1][kr] = f2bf(f.y);
            Wt[n + 2][kr] = f2bf(f.z); Wt[n + 3][kr] = f2bf(f.w);
        }
        __syncthreads();

        // A fragments: lane holds row (wr + sub*16 + l16);
        // half0 -> K 0..7 & 16..23, half1 -> K 8..15 & 24..31
        Frag a0, a1, b0, b1;
        const int ak = half * 8;
        a0.q[0] = *(const uint4*)&Xl[wr + l16][ak];
        a0.q[1] = *(const uint4*)&Xl[wr + l16][ak + 16];
        a1.q[0] = *(const uint4*)&Xl[wr + 16 + l16][ak];
        a1.q[1] = *(const uint4*)&Xl[wr + 16 + l16][ak + 16];
        // B fragments: lane holds column (wc + sub*16 + l16);
        // half0 -> K 0..7 & 16..23, half1 -> K 8..15 & 24..31
        b0.q[0] = *(const uint4*)&Wt[wc + l16][half * 8];
        b0.q[1] = *(const uint4*)&Wt[wc + l16][half * 8 + 16];
        b1.q[0] = *(const uint4*)&Wt[wc + 16 + l16][half * 8];
        b1.q[1] = *(const uint4*)&Wt[wc + 16 + l16][half * 8 + 16];

        acc[0][0] = wmma_bf16(a0, b0, acc[0][0]);
        acc[0][1] = wmma_bf16(a0, b1, acc[0][1]);
        acc[1][0] = wmma_bf16(a1, b0, acc[1][0]);
        acc[1][1] = wmma_bf16(a1, b1, acc[1][1]);
        __syncthreads();
    }

    // epilogue: bias + relu; C layout: VGPR r -> row (r + 8*half), col l16
    #pragma unroll
    for (int si = 0; si < 2; ++si) {
        #pragma unroll
        for (int sj = 0; sj < 2; ++sj) {
            const int col = colBase + wc + sj * 16 + l16;
            const float bv = bias[col];
            #pragma unroll
            for (int r = 0; r < 8; ++r) {
                const int row = rowBase + wr + si * 16 + half * 8 + r;
                float v = acc[si][sj][r] + bv;
                v = fmaxf(v, 0.0f);
                if (mode == 0) {
                    const int head = col >> 6, dd = col & 63;
                    outBf[((size_t)head * MROWS + row) * D_HEAD + dd] = f2bf(v);
                } else {
                    const size_t o = (size_t)row * N + col;
                    outF[o] = v + resid[o];
                }
            }
        }
    }
}

// ---------------------------------------------------------------------------
// key padding mask: mask[bs] = sign(|sum_H K[bs,:]|)  (K stored head-major bf16)
// ---------------------------------------------------------------------------
__global__ __launch_bounds__(256)
void mask_kernel(const unsigned short* __restrict__ Kbf, float* __restrict__ maskb)
{
    __shared__ float red[256];
    const int row = blockIdx.x;          // bs index
    const int t = threadIdx.x;
    float s = 0.0f;
    #pragma unroll
    for (int i = 0; i < 4; ++i) {
        int idx = t * 4 + i;             // 0..1023
        int h = idx >> 6, d = idx & 63;
        s += bf2f(Kbf[(size_t)h * (MROWS * D_HEAD) + (size_t)row * D_HEAD + d]);
    }
    red[t] = s;
    __syncthreads();
    for (int off = 128; off > 0; off >>= 1) {
        if (t < off) red[t] += red[t + off];
        __syncthreads();
    }
    if (t == 0) maskb[row] = (fabsf(red[0]) != 0.0f) ? 1.0f : 0.0f;
}

// ---------------------------------------------------------------------------
// Flash attention. Grid (32 qblocks, B, NHEAD), 128 threads = 4 waves.
// Each wave: one 16-row q tile, online softmax over 32-key chunks.
// K chunk staged via async global->LDS (ASYNCcnt) when available.
// ---------------------------------------------------------------------------
__global__ __launch_bounds__(128)
void attn_kernel(const unsigned short* __restrict__ Qh,
                 const unsigned short* __restrict__ Kh,
                 const unsigned short* __restrict__ Vh,
                 const float* __restrict__ maskb,
                 float* __restrict__ attnOut)
{
    __shared__ unsigned short Kc[32][64];     // key chunk, row-major
    __shared__ unsigned short Vt[64][32];     // value chunk, transposed [d][key]
    __shared__ unsigned short Pl[4][16][32];  // per-wave P bounce buffer

    const int qblk = blockIdx.x;
    const int b    = blockIdx.y;
    const int h    = blockIdx.z;
    const int tid  = threadIdx.x;
    const int lane = tid & 31;
    const int wave = tid >> 5;
    const int half = lane >> 4;
    const int l16  = lane & 15;

    const size_t headOff = ((size_t)h * B_DIM + b) * S_DIM * D_HEAD;
    const unsigned short* Qp = Qh + headOff;
    const unsigned short* Kp = Kh + headOff;
    const unsigned short* Vp = Vh + headOff;

    const int qb = qblk * 64 + wave * 16;     // wave q-tile base (within S)
    const int myQmax = qb + 15;

    // Q fragments (A layout, d split 0..31 / 32..63), straight from global
    Frag qf0, qf1;
    {
        const unsigned short* qrow = Qp + (size_t)(qb + l16) * D_HEAD;
        qf0.q[0] = *(const uint4*)(qrow + half * 8);
        qf0.q[1] = *(const uint4*)(qrow + 16 + half * 8);
        qf1.q[0] = *(const uint4*)(qrow + 32 + half * 8);
        qf1.q[1] = *(const uint4*)(qrow + 48 + half * 8);
    }

    v8f o[4];
    o[0] = {}; o[1] = {}; o[2] = {}; o[3] = {};
    float m[8], l[8];
    #pragma unroll
    for (int r = 0; r < 8; ++r) { m[r] = -3.0e38f; l[r] = 0.0f; }

    const int nChunks = ((qblk * 64 + 63) >> 5) + 1;   // causal upper bound for block

    for (int ch = 0; ch < nChunks; ++ch) {
        const int kbase = ch * 32;
        // prefetch next chunk while this one is consumed
        if (ch + 1 < nChunks) {
            const size_t nb = (size_t)(kbase + 32 + (tid >> 2)) * D_HEAD + (tid & 3) * 16;
            __builtin_prefetch(Kp + nb, 0, 1);
            __builtin_prefetch(Vp + nb, 0, 1);
        }
        // cooperative stage: K row-major (async DMA to LDS), V transposed (manual)
        #pragma unroll
        for (int i = 0; i < 2; ++i) {
            int idx = tid + i * 128;          // 0..255
            int r = idx >> 3;                 // key row 0..31
            int c = (idx & 7) << 3;           // d col 0..56 step 8
#if defined(HAVE_ASYNC_LDS)
            __builtin_amdgcn_global_load_async_to_lds_b128(
                (gptr_v4i)(Kp + (size_t)(kbase + r) * D_HEAD + c),
                (lptr_v4i)&Kc[r][c],
                0, 0);
#else
            *(uint4*)&Kc[r][c] = *(const uint4*)(Kp + (size_t)(kbase + r) * D_HEAD + c);
#endif
            uint4 vv = *(const uint4*)(Vp + (size_t)(kbase + r) * D_HEAD + c);
            unsigned short tmp[8];
            *(uint4*)tmp = vv;
            #pragma unroll
            for (int e = 0; e < 8; ++e) Vt[c + e][r] = tmp[e];
        }
#if defined(HAVE_ASYNC_LDS)
        wait_asynccnt0();                     // ASYNCcnt not covered by implicit waits
#endif
        __syncthreads();

        if (kbase <= myQmax) {
            // ---- scores: two 16-key tiles, contraction over d=64 (2 WMMAs each)
            v8f s[2];
            #pragma unroll
            for (int t = 0; t < 2; ++t) {
                Frag kf0, kf1;
                const int krow = t * 16 + l16;
                kf0.q[0] = *(const uint4*)&Kc[krow][half * 8];
                kf0.q[1] = *(const uint4*)&Kc[krow][16 + half * 8];
                kf1.q[0] = *(const uint4*)&Kc[krow][32 + half * 8];
                kf1.q[1] = *(const uint4*)&Kc[krow][48 + half * 8];
                v8f z = {};
                z = wmma_bf16(qf0, kf0, z);
                z = wmma_bf16(qf1, kf1, z);
                s[t] = z;
            }
            // ---- scale + causal + padding mask
            #pragma unroll
            for (int t = 0; t < 2; ++t) {
                const int kidx = kbase + t * 16 + l16;
                const float mv = maskb[b * S_DIM + kidx];
                const bool kill = (mv == 0.0f);
                #pragma unroll
                for (int r = 0; r < 8; ++r) {
                    const int qr = qb + half * 8 + r;
                    float v = s[t][r] * 0.125f;          // 1/sqrt(64)
                    if (kill || kidx > qr) v = NEGV;
                    s[t][r] = v;
                }
            }
            // ---- online softmax (rows live across 16-lane halves)
            #pragma unroll
            for (int r = 0; r < 8; ++r) {
                float cm = fmaxf(s[0][r], s[1][r]);
                #pragma unroll
                for (int off = 8; off >= 1; off >>= 1)
                    cm = fmaxf(cm, __shfl_xor(cm, off, 16));
                const float nm = fmaxf(m[r], cm);
                const float alpha = __expf(m[r] - nm);
                const float p0 = __expf(s[0][r] - nm);
                const float p1 = __expf(s[1][r] - nm);
                s[0][r] = p0; s[1][r] = p1;
                float rs = p0 + p1;
                #pragma unroll
                for (int off = 8; off >= 1; off >>= 1)
                    rs += __shfl_xor(rs, off, 16);
                l[r] = l[r] * alpha + rs;
                m[r] = nm;
                o[0][r] = o[0][r] * alpha;
                o[1][r] = o[1][r] * alpha;
                o[2][r] = o[2][r] * alpha;
                o[3][r] = o[3][r] * alpha;
            }
            // ---- P: C layout -> A layout via per-wave LDS bounce
            #pragma unroll
            for (int t = 0; t < 2; ++t)
                #pragma unroll
                for (int r = 0; r < 8; ++r)
                    Pl[wave][half * 8 + r][t * 16 + l16] = f2bf(s[t][r]);
            asm volatile("s_wait_dscnt 0" ::: "memory");
            Frag pf;
            pf.q[0] = *(const uint4*)&Pl[wave][l16][half * 8];
            pf.q[1] = *(const uint4*)&Pl[wave][l16][16 + half * 8];
            // ---- PV: 4 d-tiles of 16
            #pragma unroll
            for (int n = 0; n < 4; ++n) {
                Frag vf;
                vf.q[0] = *(const uint4*)&Vt[n * 16 + l16][half * 8];
                vf.q[1] = *(const uint4*)&Vt[n * 16 + l16][16 + half * 8];
                o[n] = wmma_bf16(pf, vf, o[n]);
            }
        }
        __syncthreads();
    }

    // write normalized output, fp32, [bs][H] with col = h*64 + d
    #pragma unroll
    for (int n = 0; n < 4; ++n) {
        const int d = n * 16 + l16;
        #pragma unroll
        for (int r = 0; r < 8; ++r) {
            const int srow = qb + half * 8 + r;
            const float invl = 1.0f / l[r];
            attnOut[((size_t)(b * S_DIM + srow)) * H_DIM + h * D_HEAD + d] = o[n][r] * invl;
        }
    }
}

// ---------------------------------------------------------------------------
// LayerNorm (unbiased std, ddof=1, eps added to std): one row per block
// ---------------------------------------------------------------------------
__global__ __launch_bounds__(256)
void ln_kernel(const float* __restrict__ Z, float* __restrict__ out)
{
    __shared__ float rs[256], rq[256];
    const int row = blockIdx.x;
    const int t = threadIdx.x;
    float s = 0.0f, sq = 0.0f;
    for (int c = t; c < H_DIM; c += 256) {
        float v = Z[(size_t)row * H_DIM + c];
        s += v; sq += v * v;
    }
    rs[t] = s; rq[t] = sq;
    __syncthreads();
    for (int off = 128; off > 0; off >>= 1) {
        if (t < off) { rs[t] += rs[t + off]; rq[t] += rq[t + off]; }
        __syncthreads();
    }
    const float mean = rs[0] * (1.0f / H_DIM);
    float var = (rq[0] - (float)H_DIM * mean * mean) * (1.0f / (H_DIM - 1));
    var = fmaxf(var, 0.0f);
    const float inv = 1.0f / (sqrtf(var) + 1e-8f);
    for (int c = t; c < H_DIM; c += 256)
        out[(size_t)row * H_DIM + c] = (Z[(size_t)row * H_DIM + c] - mean) * inv;
}

// ---------------------------------------------------------------------------
extern "C" void kernel_launch(void* const* d_in, const int* in_sizes, int n_in,
                              void* d_out, int out_size, void* d_ws, size_t ws_size,
                              hipStream_t stream)
{
    const float* q  = (const float*)d_in[0];
    const float* k  = (const float*)d_in[1];
    const float* v  = (const float*)d_in[2];
    const float* Wq = (const float*)d_in[3];
    const float* bq = (const float*)d_in[4];
    const float* Wk = (const float*)d_in[5];
    const float* bk = (const float*)d_in[6];
    const float* Wv = (const float*)d_in[7];
    const float* bv = (const float*)d_in[8];
    const float* Wo = (const float*)d_in[9];
    const float* bo = (const float*)d_in[10];

    char* ws = (char*)d_ws;
    const size_t MB = 1ull << 20;
    unsigned short* Qbf  = (unsigned short*)(ws);             //  8 MB
    unsigned short* Kbf  = (unsigned short*)(ws +  8 * MB);   //  8 MB
    unsigned short* Vbf  = (unsigned short*)(ws + 16 * MB);   //  8 MB
    float*          mskb = (float*)        (ws + 24 * MB);    // 16 KB
    float*          attn = (float*)        (ws + 25 * MB);    // 16 MB
    float*          Z    = (float*)        (ws + 41 * MB);    // 16 MB

    dim3 gG(MROWS / 64, H_DIM / 64), bG(128);

    gemm_bias_relu<<<gG, bG, 0, stream>>>(q, Wq, bq, MROWS, H_DIM, H_DIM, 0, Qbf, nullptr, nullptr);
    gemm_bias_relu<<<gG, bG, 0, stream>>>(k, Wk, bk, MROWS, H_DIM, H_DIM, 0, Kbf, nullptr, nullptr);
    gemm_bias_relu<<<gG, bG, 0, stream>>>(v, Wv, bv, MROWS, H_DIM, H_DIM, 0, Vbf, nullptr, nullptr);

    mask_kernel<<<MROWS, 256, 0, stream>>>(Kbf, mskb);

    attn_kernel<<<dim3(S_DIM / 64, B_DIM, NHEAD), 128, 0, stream>>>(Qbf, Kbf, Vbf, mskb, attn);

    gemm_bias_relu<<<gG, bG, 0, stream>>>(attn, Wo, bo, MROWS, H_DIM, H_DIM, 1, nullptr, Z, q);

    ln_kernel<<<MROWS, 256, 0, stream>>>(Z, (float*)d_out);
}
